// Model_77841987273136
// MI455X (gfx1250) — matrix operand order repaired and checked by reading, not compiled, not run on hardware
//
#include <hip/hip_runtime.h>
#include <cstdint>

// ---------------- problem constants ----------------
#define DD    128
#define PP    8192
#define AAN   4096          // AND (or NOT) nodes per level
#define LLV   8
#define NL    (LLV * PP)    // 65536 total nodes
#define E_AND 8192
#define E_NOT 4096
#define EPL   (E_AND + E_NOT)

typedef __attribute__((ext_vector_type(16))) _Float16 v16h;
typedef __attribute__((ext_vector_type(8)))  float    v8f;

union AB16 { uint32_t u[8]; v16h v; };

__device__ __forceinline__ void put4(AB16& f, int i, uint4 q) {
    f.u[i + 0] = q.x; f.u[i + 1] = q.y; f.u[i + 2] = q.z; f.u[i + 3] = q.w;
}

__device__ __forceinline__ float sigmoidf_(float x) {
    return 1.0f / (1.0f + __expf(-x));
}

// ---------------------------------------------------------------------------
// WMMA GEMM:  C[M,N] = act( A[M,K] (f16, row-major) * Btiled + bias )
// Btiled layout: [K/32][N][16] dwords; entry (kb, col, kl) = f16 pair
//   (B[kb*32 + 2*kl][col], B[kb*32 + 2*kl + 1][col]).
// Fragments are 2 contiguous b128 loads per lane for both A and B.
// Double-buffered K pipeline: loads for slice kb+1 are issued before the
// 4 WMMAs of slice kb, so VMEM latency is hidden under the matrix pipe.
// One wave -> 16x64 tile; blockDim = 128 (4 waves); grid = (M/64, N/64).
// ---------------------------------------------------------------------------
__device__ __forceinline__ void load_frag(
    const uint4* __restrict__ A4base, const uint4* __restrict__ B4base,
    int bStepK, int hf, int kb, AB16& a, AB16 (&b)[4])
{
    const uint4* A4 = A4base + (kb << 2);          // (kb*32 halfwords)/8
    put4(a, 0, A4[2 * hf]);                        // halfwords kb*32 + hf*16 + {0..7}
    put4(a, 4, A4[2 * hf + 1]);                    // halfwords kb*32 + hf*16 + 8 + {0..7}
    const uint4* B4 = B4base + (size_t)kb * bStepK;
#pragma unroll
    for (int t = 0; t < 4; ++t) {
        put4(b[t], 0, B4[t * 64]);                 // dwords hf*8 + {0..3}
        put4(b[t], 4, B4[t * 64 + 1]);             // dwords hf*8 + {4..7}
    }
}

__device__ __forceinline__ void mma4(v8f (&acc)[4], const AB16& a, const AB16 (&b)[4])
{
#pragma unroll
    for (int t = 0; t < 4; ++t)
        acc[t] = __builtin_amdgcn_wmma_f32_16x16x32_f16(
            false, a.v, false, b[t].v, (short)0, acc[t], false, false);
}

__global__ __launch_bounds__(128) void gemm_wmma(
    const _Float16* __restrict__ A, const uint32_t* __restrict__ Bt,
    const float* __restrict__ bias, _Float16* __restrict__ outH,
    float* __restrict__ outF, int M, int N, int K, int relu)
{
    const int lane = threadIdx.x & 31;
    const int wave = threadIdx.x >> 5;
    const int r    = lane & 15;
    const int hf   = lane >> 4;
    const int row0 = (blockIdx.x * 4 + wave) * 16;
    const int col0 = blockIdx.y * 64;

    const v8f zero = {0.f, 0.f, 0.f, 0.f, 0.f, 0.f, 0.f, 0.f};
    v8f acc[4] = {zero, zero, zero, zero};

    const uint4* A4base = reinterpret_cast<const uint4*>(A + (size_t)(row0 + r) * K);
    const uint4* B4base = reinterpret_cast<const uint4*>(Bt)
                        + (((size_t)(col0 + r)) << 2) + (hf << 1);
    const int bStepK = N << 2;                     // uint4s per 32-K block
    const int KB = K >> 5;                         // 4 or 8 (always even)

    AB16 a0, a1;
    AB16 b0[4], b1[4];
    load_frag(A4base, B4base, bStepK, hf, 0, a0, b0);

    for (int kb = 0; kb < KB; kb += 2) {
        // issue next slice's loads before consuming the current one
        load_frag(A4base, B4base, bStepK, hf, kb + 1, a1, b1);
        if (kb + 2 < KB)
            __builtin_prefetch(reinterpret_cast<const char*>(A4base) + (kb + 2) * 64, 0, 3);
        mma4(acc, a0, b0);
        if (kb + 2 < KB)
            load_frag(A4base, B4base, bStepK, hf, kb + 2, a0, b0);
        mma4(acc, a1, b1);
    }

#pragma unroll
    for (int t = 0; t < 4; ++t) {
        const int col = col0 + t * 16 + r;
        const float bv = bias ? bias[col] : 0.0f;
#pragma unroll
        for (int rr = 0; rr < 8; ++rr) {
            float val = acc[t][rr] + bv;
            if (relu) val = fmaxf(val, 0.0f);
            const size_t o = (size_t)(row0 + rr + hf * 8) * N + col;
            if (outH) outH[o] = (_Float16)val;
            if (outF) outF[o] = val;
        }
    }
}

// -------- weight fp32 -> f16, pre-tiled B layout [K/32][N][16] dwords ------
__global__ void pack_w(const float* __restrict__ W, uint32_t* __restrict__ Bt,
                       int K, int N)
{
    const int idx = blockIdx.x * blockDim.x + threadIdx.x;
    if (idx >= (K / 2) * N) return;
    const int kp = idx / N, col = idx - kp * N;
    const int kb = kp >> 4, kl = kp & 15;
    union { _Float16 f[2]; uint32_t u; } x;
    x.f[0] = (_Float16)W[(size_t)(2 * kp) * N + col];
    x.f[1] = (_Float16)W[(size_t)(2 * kp + 1) * N + col];
    Bt[(((size_t)kb * N + col) << 4) + kl] = x.u;
}

// ---------------- state init: hs = hs_init, hf = 0 (rows [0,P)) ------------
__global__ void init_state(const float* __restrict__ hsInit,
                           float* __restrict__ hsF, float* __restrict__ hfF,
                           _Float16* __restrict__ hs16, _Float16* __restrict__ hf16,
                           int n)
{
    const int i = blockIdx.x * blockDim.x + threadIdx.x;
    if (i >= n) return;
    const float v = hsInit[i];
    hsF[i]  = v;             hfF[i]  = 0.0f;
    hs16[i] = (_Float16)v;   hf16[i] = (_Float16)0.0f;
}

// ---------------- edge gathers (rows of 128 f16 = 16 uint4) ----------------
__global__ void gather_rows(const uint4* __restrict__ src, const int* __restrict__ idx,
                            uint4* __restrict__ out, int nE)
{
    const int t = blockIdx.x * blockDim.x + threadIdx.x;
    if (t >= nE * 16) return;
    const int e = t >> 4, c = t & 15;
    out[t] = src[(size_t)idx[e] * 16 + c];
}

__global__ void gather_rows2(const uint4* __restrict__ hs, const uint4* __restrict__ hf,
                             const int* __restrict__ idx, uint4* __restrict__ out, int nE)
{
    const int t = blockIdx.x * blockDim.x + threadIdx.x;
    if (t >= nE * 32) return;
    const int e = t >> 5, c = t & 31;
    out[t] = (c < 16) ? hs[(size_t)idx[e] * 16 + c]
                      : hf[(size_t)idx[e] * 16 + (c - 16)];
}

// ------- AND attention: 2 edges/node, softmax over pair (shift-invariant) --
__global__ void attn_and(const _Float16* __restrict__ Mh,   // [2*nodes,128]
                         const float* __restrict__ w2,      // attn_w[128:256]
                         _Float16* __restrict__ out, int nodes)
{
    const int node = blockIdx.x * (blockDim.x >> 5) + (threadIdx.x >> 5);
    const int lane = threadIdx.x & 31;
    if (node >= nodes) return;
    const _Float16* m0 = Mh + (size_t)(2 * node) * DD;
    const _Float16* m1 = m0 + DD;
    float v0[4], v1[4];
    float s0 = 0.f, s1 = 0.f;
#pragma unroll
    for (int c = 0; c < 4; ++c) {
        const int f = lane * 4 + c;
        v0[c] = (float)m0[f];  v1[c] = (float)m1[f];
        const float w = w2[f];
        s0 += v0[c] * w;       s1 += v1[c] * w;
    }
#pragma unroll
    for (int off = 16; off > 0; off >>= 1) {
        s0 += __shfl_xor(s0, off, 32);
        s1 += __shfl_xor(s1, off, 32);
    }
    const float mx = fmaxf(s0, s1);
    const float e0 = __expf(s0 - mx), e1 = __expf(s1 - mx);
    const float a0 = e0 / (e0 + e1), a1 = 1.0f - a0;
#pragma unroll
    for (int c = 0; c < 4; ++c) {
        const int f = lane * 4 + c;
        out[(size_t)node * DD + f] = (_Float16)(a0 * v0[c] + a1 * v1[c]);
    }
}

// ------- GRU with h==0:  r=sig(gi_r+bhh_r), z=sig(gi_z+bhh_z),
//         n=tanh(gi_n + r*bhh_n), h' = (1-z)*n ------------------------------
__global__ void gru_ew(const float* __restrict__ gi, const float* __restrict__ bhh,
                       float* __restrict__ hF, _Float16* __restrict__ h16, int rows)
{
    const int idx = blockIdx.x * blockDim.x + threadIdx.x;
    if (idx >= rows * DD) return;
    const int i = idx >> 7, j = idx & 127;
    const float* g = gi + (size_t)i * 384;
    const float r = sigmoidf_(g[j]       + bhh[j]);
    const float z = sigmoidf_(g[128 + j] + bhh[128 + j]);
    const float n = tanhf    (g[256 + j] + r * bhh[256 + j]);
    const float h = (1.0f - z) * n;
    hF[idx]  = h;
    h16[idx] = (_Float16)h;
}

// ===========================================================================
extern "C" void kernel_launch(void* const* d_in, const int* in_sizes, int n_in,
                              void* d_out, int out_size, void* d_ws, size_t ws_size,
                              hipStream_t stream)
{
    (void)out_size; (void)ws_size;
    // internal order for aggr/gru slots: 0=and_strc 1=and_func 2=not_strc 3=not_func
    const float* msgW[4][3]; const float* msgB[4][3]; const float* attnW[4];
    const float* Wih[4]; const float* bih[4]; const float* bhh[4];
    const float* hsInit = nullptr; const int* esrc = nullptr;
    auto F = [&](int i) { return (const float*)d_in[i]; };

    if (n_in >= 3 && in_sizes[0] == EPL * (LLV - 1)) {
        // fully key-sorted flatten: edge_dst, edge_src, hs_init, params(sorted)
        esrc = (const int*)d_in[1];
        hsInit = F(2);
        const int pos[4] = {1, 0, 3, 2};   // internal -> sorted position
        for (int a = 0; a < 4; ++a) {
            const int b = 3 + 14 * pos[a];
            attnW[a] = F(b + 1);
            for (int j = 0; j < 3; ++j) { msgW[a][j] = F(b + 2 + 2 * j); msgB[a][j] = F(b + 3 + 2 * j); }
        }
        for (int g = 0; g < 4; ++g) {
            const int b = 3 + 56 + 4 * pos[g];           // W_hh, W_ih, b_hh, b_ih
            Wih[g] = F(b + 1); bhh[g] = F(b + 2); bih[g] = F(b + 3);
        }
    } else if (n_in >= 3 && in_sizes[1] == 1) {
        // top-level insertion order, nested key-sorted
        hsInit = F(0);
        esrc = (const int*)d_in[73];
        const int pos[4] = {1, 0, 3, 2};
        for (int a = 0; a < 4; ++a) {
            const int b = 1 + 14 * pos[a];
            attnW[a] = F(b + 1);
            for (int j = 0; j < 3; ++j) { msgW[a][j] = F(b + 2 + 2 * j); msgB[a][j] = F(b + 3 + 2 * j); }
        }
        for (int g = 0; g < 4; ++g) {
            const int b = 57 + 4 * pos[g];
            Wih[g] = F(b + 1); bhh[g] = F(b + 2); bih[g] = F(b + 3);
        }
    } else {
        // full insertion order: hs_init, params(msg W/b x3, q W/b x3, attn_w, attn_b),
        // gru(W_ih, W_hh, b_ih, b_hh), edge_src, edge_dst
        hsInit = F(0);
        esrc = (const int*)d_in[73];
        for (int a = 0; a < 4; ++a) {
            const int b = 1 + 14 * a;
            for (int j = 0; j < 3; ++j) { msgW[a][j] = F(b + 2 * j); msgB[a][j] = F(b + 1 + 2 * j); }
            attnW[a] = F(b + 12);
        }
        for (int g = 0; g < 4; ++g) {
            const int b = 57 + 4 * g;
            Wih[g] = F(b + 0); bih[g] = F(b + 2); bhh[g] = F(b + 3);
        }
    }

    // ---------------- workspace layout ----------------
    char* p = (char*)d_ws;
    auto walloc = [&](size_t bytes) -> void* {
        void* r = (void*)p; p += (bytes + 255) & ~(size_t)255; return r;
    };
    _Float16* hs16 = (_Float16*)walloc((size_t)NL * DD * 2);
    _Float16* hf16 = (_Float16*)walloc((size_t)NL * DD * 2);
    _Float16* Xe   = (_Float16*)walloc((size_t)E_AND * 256 * 2);
    _Float16* T1   = (_Float16*)walloc((size_t)E_AND * DD * 2);
    _Float16* T2   = (_Float16*)walloc((size_t)E_AND * DD * 2);
    _Float16* Mb   = (_Float16*)walloc((size_t)E_AND * DD * 2);
    _Float16* msgN = (_Float16*)walloc((size_t)AAN * DD * 2);
    float*    gi   = (float*)   walloc((size_t)AAN * 384 * 4);
    uint32_t* msgWp[4][3]; uint32_t* WihP[4];
    for (int a = 0; a < 4; ++a)
        for (int j = 0; j < 3; ++j) {
            const int K = (a == 1 && j == 0) ? 256 : DD;
            msgWp[a][j] = (uint32_t*)walloc((size_t)(K / 2) * DD * 4);
        }
    for (int g = 0; g < 4; ++g) WihP[g] = (uint32_t*)walloc((size_t)64 * 384 * 4);

    // ---------------- pack weights to tiled f16 B-layout ----------------
    auto pack = [&](const float* W, uint32_t* Bt, int K, int N) {
        const int tot = (K / 2) * N;
        pack_w<<<(tot + 255) / 256, 256, 0, stream>>>(W, Bt, K, N);
    };
    for (int a = 0; a < 4; ++a)
        for (int j = 0; j < 3; ++j)
            pack(msgW[a][j], msgWp[a][j], (a == 1 && j == 0) ? 256 : DD, DD);
    for (int g = 0; g < 4; ++g) pack(Wih[g], WihP[g], DD, 384);

    // ---------------- init masters (fp32 in d_out) + f16 shadows ----------
    float* hsF = (float*)d_out;
    float* hfF = hsF + (size_t)NL * DD;
    {
        const int tot = PP * DD;
        init_state<<<(tot + 255) / 256, 256, 0, stream>>>(hsInit, hsF, hfF, hs16, hf16, tot);
    }

    auto gemm = [&](const _Float16* A, const uint32_t* Bt, const float* bias,
                    _Float16* oH, float* oF, int M, int N, int K, int relu) {
        dim3 grid(M / 64, N / 64);
        gemm_wmma<<<grid, 128, 0, stream>>>(A, Bt, bias, oH, oF, M, N, K, relu);
    };

    // ---------------- level-by-level pipeline ----------------
    for (int l = 1; l < LLV; ++l) {
        const int n0 = l * PP;
        const int* asrc = esrc + (size_t)(l - 1) * EPL;
        const int* nsrc = asrc + E_AND;

        // ---- AND / structural (hs) ----
        gather_rows<<<(E_AND * 16) / 256, 256, 0, stream>>>(
            (const uint4*)hs16, asrc, (uint4*)Xe, E_AND);
        gemm(Xe, msgWp[0][0], msgB[0][0], T1, nullptr, E_AND, DD, DD, 1);
        gemm(T1, msgWp[0][1], msgB[0][1], T2, nullptr, E_AND, DD, DD, 1);
        gemm(T2, msgWp[0][2], msgB[0][2], Mb, nullptr, E_AND, DD, DD, 0);
        attn_and<<<AAN / 8, 256, 0, stream>>>(Mb, attnW[0] + DD, msgN, AAN);
        gemm(msgN, WihP[0], bih[0], nullptr, gi, AAN, 384, DD, 0);
        gru_ew<<<(AAN * DD) / 256, 256, 0, stream>>>(
            gi, bhh[0], hsF + (size_t)n0 * DD, hs16 + (size_t)n0 * DD, AAN);

        // ---- AND / functional (node_state = [hs, hf]) ----
        gather_rows2<<<(E_AND * 32) / 256, 256, 0, stream>>>(
            (const uint4*)hs16, (const uint4*)hf16, asrc, (uint4*)Xe, E_AND);
        gemm(Xe, msgWp[1][0], msgB[1][0], T1, nullptr, E_AND, DD, 256, 1);
        gemm(T1, msgWp[1][1], msgB[1][1], T2, nullptr, E_AND, DD, DD, 1);
        gemm(T2, msgWp[1][2], msgB[1][2], Mb, nullptr, E_AND, DD, DD, 0);
        attn_and<<<AAN / 8, 256, 0, stream>>>(Mb, attnW[1] + DD, msgN, AAN);
        gemm(msgN, WihP[1], bih[1], nullptr, gi, AAN, 384, DD, 0);
        gru_ew<<<(AAN * DD) / 256, 256, 0, stream>>>(
            gi, bhh[1], hfF + (size_t)n0 * DD, hf16 + (size_t)n0 * DD, AAN);

        // ---- NOT / structural (hs); single edge -> alpha == 1 ----
        gather_rows<<<(E_NOT * 16) / 256, 256, 0, stream>>>(
            (const uint4*)hs16, nsrc, (uint4*)Xe, E_NOT);
        gemm(Xe, msgWp[2][0], msgB[2][0], T1, nullptr, E_NOT, DD, DD, 1);
        gemm(T1, msgWp[2][1], msgB[2][1], T2, nullptr, E_NOT, DD, DD, 1);
        gemm(T2, msgWp[2][2], msgB[2][2], Mb, nullptr, E_NOT, DD, DD, 0);
        gemm(Mb, WihP[2], bih[2], nullptr, gi, E_NOT, 384, DD, 0);
        gru_ew<<<(AAN * DD) / 256, 256, 0, stream>>>(
            gi, bhh[2], hsF + (size_t)(n0 + AAN) * DD, hs16 + (size_t)(n0 + AAN) * DD, AAN);

        // ---- NOT / functional (hf) ----
        gather_rows<<<(E_NOT * 16) / 256, 256, 0, stream>>>(
            (const uint4*)hf16, nsrc, (uint4*)Xe, E_NOT);
        gemm(Xe, msgWp[3][0], msgB[3][0], T1, nullptr, E_NOT, DD, DD, 1);
        gemm(T1, msgWp[3][1], msgB[3][1], T2, nullptr, E_NOT, DD, DD, 1);
        gemm(T2, msgWp[3][2], msgB[3][2], Mb, nullptr, E_NOT, DD, DD, 0);
        gemm(Mb, WihP[3], bih[3], nullptr, gi, E_NOT, 384, DD, 0);
        gru_ew<<<(AAN * DD) / 256, 256, 0, stream>>>(
            gi, bhh[3], hfF + (size_t)(n0 + AAN) * DD, hf16 + (size_t)(n0 + AAN) * DD, AAN);
    }
}
